// EncoderBlock_89043261981132
// MI455X (gfx1250) — matrix operand order repaired and checked by reading, not compiled, not run on hardware
//
#include <hip/hip_runtime.h>
#include <hip/hip_bf16.h>

typedef __attribute__((ext_vector_type(16))) _Float16 v16h;
typedef __attribute__((ext_vector_type(8)))  _Float16 v8h;
typedef __attribute__((ext_vector_type(8)))  float    v8f;
typedef __attribute__((ext_vector_type(4)))  float    v4f;

#define NATOMS 50000
#define NEDGES 1000000
#define DF 32

__device__ __forceinline__ int lane_id() { return (int)(threadIdx.x & 31u); }

// ---- WMMA fragment helpers (CDNA5 ISA 7.12.2 layouts, f16 in / f32 acc) ----

// A: 16x32 (MxK). lane<16: row=lane, K {0..7}U{16..23}; lane>=16: row=lane&15, K {8..15}U{24..31}
// f32 source (global or LDS), 16B vector loads, convert to f16.
__device__ __forceinline__ v16h load_A_f32(const float* __restrict__ src, int ld) {
  int l = lane_id();
  int m = l & 15;
  int kofs = (l < 16) ? 0 : 8;
  const float* row = src + m * ld + kofs;
  v4f x0 = *(const v4f*)(row);
  v4f x1 = *(const v4f*)(row + 4);
  v4f x2 = *(const v4f*)(row + 16);
  v4f x3 = *(const v4f*)(row + 20);
  v16h a;
#pragma unroll
  for (int i = 0; i < 4; ++i) {
    a[i]      = (_Float16)x0[i];
    a[4 + i]  = (_Float16)x1[i];
    a[8 + i]  = (_Float16)x2[i];
    a[12 + i] = (_Float16)x3[i];
  }
  return a;
}

// f16 source (LDS staging), two 16B vector loads.
__device__ __forceinline__ v16h load_A_f16(const _Float16* __restrict__ src, int ld) {
  int l = lane_id();
  int m = l & 15;
  int kofs = (l < 16) ? 0 : 8;
  const _Float16* row = src + m * ld + kofs;
  v8h x0 = *(const v8h*)(row);
  v8h x1 = *(const v8h*)(row + 16);
  v16h a;
#pragma unroll
  for (int i = 0; i < 8; ++i) { a[i] = x0[i]; a[8 + i] = x1[i]; }
  return a;
}

// B fragments pre-packed as [frag][lane][16] f16 -> one 32B per-lane load.
__device__ __forceinline__ v16h load_B_packed(const _Float16* __restrict__ p, int frag) {
  return *((const v16h*)p + (size_t)frag * 32 + lane_id());
}

// C/D: 16x16 f32, lane<16: n=lane, rows 0..7; lane>=16: n=lane-16, rows 8..15
__device__ __forceinline__ v8f c_init_bias(const float* __restrict__ bias_n0) {
  float bv = bias_n0[lane_id() & 15];
  v8f c;
#pragma unroll
  for (int i = 0; i < 8; ++i) c[i] = bv;
  return c;
}

__device__ __forceinline__ void store_C_f32(float* __restrict__ dst, int ld, v8f c, bool relu) {
  int l = lane_id();
  int n = l & 15;
  int mofs = (l < 16) ? 0 : 8;
#pragma unroll
  for (int i = 0; i < 8; ++i) {
    float x = c[i];
    if (relu) x = fmaxf(x, 0.0f);
    dst[(mofs + i) * ld + n] = x;
  }
}

__device__ __forceinline__ void store_C_f16(_Float16* __restrict__ dst, int ld, v8f c, bool relu) {
  int l = lane_id();
  int n = l & 15;
  int mofs = (l < 16) ? 0 : 8;
#pragma unroll
  for (int i = 0; i < 8; ++i) {
    float x = c[i];
    if (relu) x = fmaxf(x, 0.0f);
    dst[(mofs + i) * ld + n] = (_Float16)x;
  }
}

__device__ __forceinline__ v8f wmma16(v16h a, v16h b, v8f c) {
  return __builtin_amdgcn_wmma_f32_16x16x32_f16(false, a, false, b, (short)0, c, false, false);
}

// -------------------- kernels --------------------

// Pack one weight matrix [K x N] f32 into f16 B-fragments: frag = kt*ntiles + nt.
__global__ void k_pack_w(const float* __restrict__ W, int N, int ntiles,
                         _Float16* __restrict__ out) {
  int frag = blockIdx.x;
  int kt = frag / ntiles, nt = frag - kt * ntiles;
  int l = threadIdx.x & 31;
  int n = l & 15;
  int kbase = (l < 16) ? 0 : 8;
  const float* Wt = W + (size_t)(kt * 32) * N + nt * 16 + n;
  v16h b;
#pragma unroll
  for (int i = 0; i < 16; ++i) b[i] = (_Float16)Wt[(size_t)(kbase + i + (i >= 8 ? 8 : 0)) * N];
  *((v16h*)out + (size_t)frag * 32 + l) = b;
}

__global__ void k_init(float* agg, float* amax, float* denom) {
  int i = blockIdx.x * blockDim.x + threadIdx.x;
  if (i < NATOMS * DF) agg[i] = 0.0f;
  if (i < NATOMS) { amax[i] = -__builtin_inff(); denom[i] = 0.0f; }
}

// q/k/v/sa = atoms @ {Wq,Wk,Wv,Ws} + bias   (one wave per 16-row tile)
__global__ void k_node_linears(const float* __restrict__ atoms,
                               const _Float16* __restrict__ pWq, const float* __restrict__ bq,
                               const _Float16* __restrict__ pWk, const float* __restrict__ bk,
                               const _Float16* __restrict__ pWv, const float* __restrict__ bv,
                               const _Float16* __restrict__ pWs, const float* __restrict__ bs,
                               float* __restrict__ q, float* __restrict__ k,
                               float* __restrict__ v, float* __restrict__ sa, int ntiles) {
  int tile = blockIdx.x * (blockDim.x >> 5) + (int)(threadIdx.x >> 5);
  if (tile >= ntiles) return;
  v16h A = load_A_f32(atoms + (size_t)tile * 16 * DF, DF);
  const _Float16* Wt[4] = {pWq, pWk, pWv, pWs};
  const float* bt[4] = {bq, bk, bv, bs};
  float* ot[4] = {q, k, v, sa};
#pragma unroll
  for (int w = 0; w < 4; ++w) {
#pragma unroll
    for (int nt = 0; nt < 2; ++nt) {
      v16h B = load_B_packed(Wt[w], nt);
      v8f C = c_init_bias(bt[w] + nt * 16);
      C = wmma16(A, B, C);
      store_C_f32(ot[w] + (size_t)tile * 16 * DF + nt * 16, DF, C, false);
    }
  }
}

// elin = bonds @ We + be
__global__ void k_edge_lin(const float* __restrict__ bonds,
                           const _Float16* __restrict__ pWe, const float* __restrict__ be,
                           float* __restrict__ elin, int ntiles) {
  int tile = blockIdx.x * (blockDim.x >> 5) + (int)(threadIdx.x >> 5);
  if (tile >= ntiles) return;
  v16h A = load_A_f32(bonds + (size_t)tile * 16 * DF, DF);
#pragma unroll
  for (int nt = 0; nt < 2; ++nt) {
    v16h B = load_B_packed(pWe, nt);
    v8f C = c_init_bias(be + nt * 16);
    C = wmma16(A, B, C);
    store_C_f32(elin + (size_t)tile * 16 * DF + nt * 16, DF, C, false);
  }
}

// alpha = scale * <q[dst], k[src]+elin>;  amax[dst] = max  (one wave per edge)
__global__ void k_edge_alpha(const float* __restrict__ q, const float* __restrict__ kbuf,
                             const float* __restrict__ elin,
                             const int* __restrict__ src, const int* __restrict__ dst,
                             float* __restrict__ alpha, float* __restrict__ amax) {
  int e = blockIdx.x * 8 + (int)(threadIdx.x >> 5);
  int l = threadIdx.x & 31;
  if (e >= NEDGES) return;
  int s = src[e], d = dst[e];
  float kv = kbuf[(size_t)s * DF + l] + elin[(size_t)e * DF + l];
  float p = q[(size_t)d * DF + l] * kv;
#pragma unroll
  for (int off = 16; off > 0; off >>= 1) p += __shfl_xor(p, off, 32);
  if (l == 0) {
    float a = p * 0.17677669529663687f;  // 1/sqrt(32)
    alpha[e] = a;
    // float atomic max via monotone int/uint ordering (amax pre-init to -inf)
    if (a >= 0.0f) atomicMax((int*)(amax + d), __float_as_int(a));
    else atomicMin((unsigned int*)(amax + d), __float_as_uint(a));
  }
}

// ex = exp(alpha - amax[dst]); denom[dst] += ex  (alpha overwritten with ex)
__global__ void k_edge_expsum(float* __restrict__ alpha, const float* __restrict__ amax,
                              const int* __restrict__ dst, float* __restrict__ denom) {
  int e = blockIdx.x * blockDim.x + threadIdx.x;
  if (e >= NEDGES) return;
  int d = dst[e];
  float ex = __expf(alpha[e] - amax[d]);
  alpha[e] = ex;
  atomicAdd(denom + d, ex);
}

// agg[dst] += (ex/denom[dst]) * (v[src]+elin)   (one wave per edge, lane=feature)
__global__ void k_edge_agg(const float* __restrict__ vbuf, const float* __restrict__ elin,
                           const float* __restrict__ ex, const float* __restrict__ denom,
                           const int* __restrict__ src, const int* __restrict__ dst,
                           float* __restrict__ agg) {
  int e = blockIdx.x * 8 + (int)(threadIdx.x >> 5);
  int l = threadIdx.x & 31;
  if (e >= NEDGES) return;
  int s = src[e], d = dst[e];
  float w = ex[e] / (denom[d] + 1e-16f);
  float val = w * (vbuf[(size_t)s * DF + l] + elin[(size_t)e * DF + l]);
  atomicAdd(agg + (size_t)d * DF + l, val);
}

// atoms1 = atoms + agg + sa;  atoms_out = atoms1 + relu(atoms1@a1W+a1b)@a2W+a2b
__global__ void __launch_bounds__(32) k_atom_update(
    const float* __restrict__ atoms, const float* __restrict__ agg,
    const float* __restrict__ sa,
    const _Float16* __restrict__ pa1, const float* __restrict__ a1b,
    const _Float16* __restrict__ pa2, const float* __restrict__ a2b,
    float* __restrict__ atoms_out, int ntiles) {
  __shared__ float t[16 * 32];
  __shared__ _Float16 h[16 * 64];
  int tile = blockIdx.x;
  if (tile >= ntiles) return;
  size_t base = (size_t)tile * 512;
  for (int i = threadIdx.x; i < 512; i += 32)
    t[i] = atoms[base + i] + agg[base + i] + sa[base + i];
  __syncthreads();
  v16h A = load_A_f32(t, 32);
#pragma unroll
  for (int nt = 0; nt < 4; ++nt) {
    v16h B = load_B_packed(pa1, nt);
    v8f C = c_init_bias(a1b + nt * 16);
    C = wmma16(A, B, C);
    store_C_f16(h + nt * 16, 64, C, true);
  }
  __syncthreads();
#pragma unroll
  for (int nt = 0; nt < 2; ++nt) {
    v8f C = c_init_bias(a2b + nt * 16);
#pragma unroll
    for (int kt = 0; kt < 2; ++kt) {
      v16h Ah = load_A_f16(h + kt * 32, 64);
      v16h B = load_B_packed(pa2, kt * 2 + nt);
      C = wmma16(Ah, B, C);
    }
    int l = lane_id(); int n = (l & 15) + nt * 16; int mofs = (l < 16) ? 0 : 8;
#pragma unroll
    for (int i = 0; i < 8; ++i) {
      int m = mofs + i;
      atoms_out[base + (size_t)m * 32 + n] = C[i] + t[m * 32 + n];
    }
  }
}

// Fused: cat=[atoms[src],atoms[dst],bonds] -> phi_e(96->128->64->32)+bonds -> bond_fc
__global__ void __launch_bounds__(32) k_edge_mlp(
    const float* __restrict__ atoms_out, const float* __restrict__ bonds,
    const int* __restrict__ src, const int* __restrict__ dst,
    const _Float16* __restrict__ pe1, const float* __restrict__ e1b,
    const _Float16* __restrict__ pe2, const float* __restrict__ e2b,
    const _Float16* __restrict__ pe3, const float* __restrict__ e3b,
    const _Float16* __restrict__ pb1, const float* __restrict__ b1b,
    const _Float16* __restrict__ pb2, const float* __restrict__ b2b,
    float* __restrict__ bonds_out, int ntiles) {
  __shared__ _Float16 cat[16 * 96];   // reused as 16x64 stage buffer
  __shared__ _Float16 h1[16 * 128];
  __shared__ float bm[16 * 32];       // bonds_mid residual, f32
  int tile = blockIdx.x;
  if (tile >= ntiles) return;
  int e0 = tile * 16;
  // prefetch next tile's bonds rows while we gather (global_prefetch_b8)
  if (e0 + 16 < NEDGES)
    __builtin_prefetch(bonds + (size_t)(e0 + 16) * 32 + (threadIdx.x & 31) * 16, 0, 1);
  // gather concat tile (f32 global -> f16 LDS)
  for (int idx = threadIdx.x; idx < 16 * 96; idx += 32) {
    int r = idx / 96, c = idx - r * 96;
    int e = e0 + r;
    float val;
    if (c < 32)      val = atoms_out[(size_t)src[e] * 32 + c];
    else if (c < 64) val = atoms_out[(size_t)dst[e] * 32 + (c - 32)];
    else             val = bonds[(size_t)e * 32 + (c - 64)];
    cat[idx] = (_Float16)val;
  }
  __syncthreads();
  // GEMM1: [16x96]x[96x128] -> h1, relu
#pragma unroll
  for (int nt = 0; nt < 8; ++nt) {
    v8f C = c_init_bias(e1b + nt * 16);
#pragma unroll
    for (int kt = 0; kt < 3; ++kt) {
      v16h A = load_A_f16(cat + kt * 32, 96);
      v16h B = load_B_packed(pe1, kt * 8 + nt);
      C = wmma16(A, B, C);
    }
    store_C_f16(h1 + nt * 16, 128, C, true);
  }
  __syncthreads();
  // GEMM2: [16x128]x[128x64] -> cat[0:16*64], relu
#pragma unroll
  for (int nt = 0; nt < 4; ++nt) {
    v8f C = c_init_bias(e2b + nt * 16);
#pragma unroll
    for (int kt = 0; kt < 4; ++kt) {
      v16h A = load_A_f16(h1 + kt * 32, 128);
      v16h B = load_B_packed(pe2, kt * 4 + nt);
      C = wmma16(A, B, C);
    }
    store_C_f16(cat + nt * 16, 64, C, true);
  }
  __syncthreads();
  // GEMM3: [16x64]x[64x32] + bonds -> bonds_mid (f32, LDS)
#pragma unroll
  for (int nt = 0; nt < 2; ++nt) {
    v8f C = c_init_bias(e3b + nt * 16);
#pragma unroll
    for (int kt = 0; kt < 2; ++kt) {
      v16h A = load_A_f16(cat + kt * 32, 64);
      v16h B = load_B_packed(pe3, kt * 2 + nt);
      C = wmma16(A, B, C);
    }
    int l = lane_id(); int n = (l & 15) + nt * 16; int mofs = (l < 16) ? 0 : 8;
#pragma unroll
    for (int i = 0; i < 8; ++i) {
      int m = mofs + i;
      bm[m * 32 + n] = C[i] + bonds[((size_t)(e0 + m)) * 32 + n];
    }
  }
  __syncthreads();
  // GEMM4: bonds_mid[16x32]x[32x64] -> cat, relu
  v16h Abm = load_A_f32(bm, 32);
#pragma unroll
  for (int nt = 0; nt < 4; ++nt) {
    v16h B = load_B_packed(pb1, nt);
    v8f C = c_init_bias(b1b + nt * 16);
    C = wmma16(Abm, B, C);
    store_C_f16(cat + nt * 16, 64, C, true);
  }
  __syncthreads();
  // GEMM5: [16x64]x[64x32] + bonds_mid -> bonds_out
#pragma unroll
  for (int nt = 0; nt < 2; ++nt) {
    v8f C = c_init_bias(b2b + nt * 16);
#pragma unroll
    for (int kt = 0; kt < 2; ++kt) {
      v16h A = load_A_f16(cat + kt * 32, 64);
      v16h B = load_B_packed(pb2, kt * 2 + nt);
      C = wmma16(A, B, C);
    }
    int l = lane_id(); int n = (l & 15) + nt * 16; int mofs = (l < 16) ? 0 : 8;
#pragma unroll
    for (int i = 0; i < 8; ++i) {
      int m = mofs + i;
      bonds_out[((size_t)(e0 + m)) * 32 + n] = C[i] + bm[m * 32 + n];
    }
  }
}

// -------------------- launcher --------------------

extern "C" void kernel_launch(void* const* d_in, const int* in_sizes, int n_in,
                              void* d_out, int out_size, void* d_ws, size_t ws_size,
                              hipStream_t stream) {
  const float* atoms = (const float*)d_in[0];
  const float* bonds = (const float*)d_in[1];
  const int* src = (const int*)d_in[2];   // bond_atom_1
  const int* dst = (const int*)d_in[3];   // bond_atom_2
  const float* Wq = (const float*)d_in[4];  const float* bq = (const float*)d_in[5];
  const float* Wk = (const float*)d_in[6];  const float* bk = (const float*)d_in[7];
  const float* Wv = (const float*)d_in[8];  const float* bv = (const float*)d_in[9];
  const float* We = (const float*)d_in[10]; const float* be = (const float*)d_in[11];
  const float* Ws = (const float*)d_in[12]; const float* bs = (const float*)d_in[13];
  const float* a1W = (const float*)d_in[14]; const float* a1b = (const float*)d_in[15];
  const float* a2W = (const float*)d_in[16]; const float* a2b = (const float*)d_in[17];
  const float* e1W = (const float*)d_in[18]; const float* e1b = (const float*)d_in[19];
  const float* e2W = (const float*)d_in[20]; const float* e2b = (const float*)d_in[21];
  const float* e3W = (const float*)d_in[22]; const float* e3b = (const float*)d_in[23];
  const float* b1W = (const float*)d_in[24]; const float* b1b = (const float*)d_in[25];
  const float* b2W = (const float*)d_in[26]; const float* b2b = (const float*)d_in[27];

  float* ws = (float*)d_ws;
  float* qb    = ws;                 // 1.6M
  float* kb    = ws + 1600000;
  float* vb    = ws + 3200000;
  float* sab   = ws + 4800000;
  float* agg   = ws + 6400000;
  float* amax  = ws + 8000000;       // 50k
  float* denom = ws + 8050000;       // 50k
  float* alpha = ws + 8100000;       // 1M (alpha then ex)
  float* elin  = ws + 9100000;       // 32M
  _Float16* pw = (_Float16*)(ws + 41100000);  // packed f16 weight fragments (70 x 512 halfs)

  _Float16* pWq = pw + (size_t)0  * 512;   // 2 frags
  _Float16* pWk = pw + (size_t)2  * 512;   // 2
  _Float16* pWv = pw + (size_t)4  * 512;   // 2
  _Float16* pWs = pw + (size_t)6  * 512;   // 2
  _Float16* pWe = pw + (size_t)8  * 512;   // 2
  _Float16* pa1 = pw + (size_t)10 * 512;   // 4
  _Float16* pa2 = pw + (size_t)14 * 512;   // 4
  _Float16* pe1 = pw + (size_t)18 * 512;   // 24
  _Float16* pe2 = pw + (size_t)42 * 512;   // 16
  _Float16* pe3 = pw + (size_t)58 * 512;   // 4
  _Float16* pb1 = pw + (size_t)62 * 512;   // 4
  _Float16* pb2 = pw + (size_t)66 * 512;   // 4

  float* atoms_out = (float*)d_out;
  float* bonds_out = (float*)d_out + (size_t)NATOMS * DF;

  const int atiles = NATOMS / 16;    // 3125
  const int etiles = NEDGES / 16;    // 62500

  // pack weights to f16 fragments (tiny; all L2-resident afterwards)
  k_pack_w<<<2,  32, 0, stream>>>(Wq,  32,  2, pWq);
  k_pack_w<<<2,  32, 0, stream>>>(Wk,  32,  2, pWk);
  k_pack_w<<<2,  32, 0, stream>>>(Wv,  32,  2, pWv);
  k_pack_w<<<2,  32, 0, stream>>>(Ws,  32,  2, pWs);
  k_pack_w<<<2,  32, 0, stream>>>(We,  32,  2, pWe);
  k_pack_w<<<4,  32, 0, stream>>>(a1W, 64,  4, pa1);
  k_pack_w<<<4,  32, 0, stream>>>(a2W, 32,  2, pa2);
  k_pack_w<<<24, 32, 0, stream>>>(e1W, 128, 8, pe1);
  k_pack_w<<<16, 32, 0, stream>>>(e2W, 64,  4, pe2);
  k_pack_w<<<4,  32, 0, stream>>>(e3W, 32,  2, pe3);
  k_pack_w<<<4,  32, 0, stream>>>(b1W, 64,  4, pb1);
  k_pack_w<<<4,  32, 0, stream>>>(b2W, 32,  2, pb2);

  k_init<<<(NATOMS * DF + 255) / 256, 256, 0, stream>>>(agg, amax, denom);
  k_node_linears<<<(atiles + 7) / 8, 256, 0, stream>>>(
      atoms, pWq, bq, pWk, bk, pWv, bv, pWs, bs, qb, kb, vb, sab, atiles);
  k_edge_lin<<<(etiles + 7) / 8, 256, 0, stream>>>(bonds, pWe, be, elin, etiles);
  k_edge_alpha<<<(NEDGES + 7) / 8, 256, 0, stream>>>(qb, kb, elin, src, dst, alpha, amax);
  k_edge_expsum<<<(NEDGES + 255) / 256, 256, 0, stream>>>(alpha, amax, dst, denom);
  k_edge_agg<<<(NEDGES + 7) / 8, 256, 0, stream>>>(vb, elin, alpha, denom, src, dst, agg);
  k_atom_update<<<atiles, 32, 0, stream>>>(atoms, agg, sab, pa1, a1b, pa2, a2b,
                                           atoms_out, atiles);
  k_edge_mlp<<<etiles, 32, 0, stream>>>(atoms_out, bonds, src, dst,
                                        pe1, e1b, pe2, e2b, pe3, e3b,
                                        pb1, b1b, pb2, b2b, bonds_out, etiles);
}
